// MultiHeadSelfAttention_50921132261478
// MI455X (gfx1250) — compile-verified, matrix operand order checked by reading, not run on hardware
//
#include <hip/hip_runtime.h>

// ---------------------------------------------------------------------------
// MI455X / gfx1250 multi-head self-attention.
// bf16 WMMA (v_wmma_f32_16x16x32_bf16) + TDM (tensor_load_to_lds) staging.
// ---------------------------------------------------------------------------

typedef __attribute__((ext_vector_type(16))) __bf16   v16bf;
typedef __attribute__((ext_vector_type(8)))  float    v8f;
typedef __attribute__((ext_vector_type(4)))  unsigned v4u;
typedef __attribute__((ext_vector_type(8)))  unsigned v8u;

#define N_HEADS 16
#define HIDDEN  1024
#define DIM     1024
#define HD      64
#define BATCH   2
#define SEQ     2048
#define MROWS   (BATCH * SEQ)                 // 4096 token rows
#define SCALE_Q 8.881784197001252e-16f        // HIDDEN^-5 = 2^-50 (faithful)
#define BIGNEG  (-3.402823466e38f)

// TDM LDS padding: pad_interval = 16 DWORDs, pad_amount = 4 DWORDs.
// A 32-element bf16 row segment (16 DW) is followed by a 16B pad -> row
// stride stays 16B-aligned for ds_load_b128 and spreads lanes across banks.
#define LROW32  40      // LDS elems per 32-elem row   (64B data + 16B pad)
#define LROW64  80      // LDS elems per 64-elem row   (2 segments + 2 pads)

union FragBF { v16bf v; uint4 u[2]; };

__device__ __forceinline__ __bf16 f2bf(float f) {
  union { float f; unsigned u; } c; c.f = f;
  unsigned r = (c.u + 0x7FFFu + ((c.u >> 16) & 1u)) >> 16;  // RNE
  union { unsigned short s; __bf16 b; } o; o.s = (unsigned short)r;
  return o.b;
}

// 16x32 bf16 A-frag / 32x16 B-frag loader (works for global or LDS pointers).
// p -> element [row_base + (lane&15)][k0]; lo half lanes K{0..7,16..23},
// hi half lanes K{8..15,24..31} per ISA 7.12.2.
__device__ __forceinline__ v16bf load_frag16(const __bf16* p, int hi) {
  FragBF f;
  f.u[0] = *(const uint4*)(p + hi * 8);
  f.u[1] = *(const uint4*)(p + 16 + hi * 8);
  return f.v;
}

__device__ __forceinline__ v8f wmma_bf16(v16bf a, v16bf b, v8f c) {
  return __builtin_amdgcn_wmma_f32_16x16x32_bf16(false, a, false, b,
                                                 (short)0, c, false, false);
}

#if __has_builtin(__builtin_amdgcn_s_wait_tensorcnt)
#define WAIT_TENSOR(n) __builtin_amdgcn_s_wait_tensorcnt(n)
#else
#define WAIT_TENSOR(n) asm volatile("s_wait_tensorcnt %0" ::"i"(n))
#endif

// Low 32 bits of a flat LDS address are the LDS byte offset (aperture in
// addr[63:32], ISA 10.2 per-aperture calc).
__device__ __forceinline__ unsigned lds_off(const void* p) {
  return (unsigned)(unsigned long long)p;
}

// TDM 2D tile load: global (row-major, bf16) -> LDS, with row padding.
// Descriptor per ISA 8.3/8.4. Groups 2/3 omitted (2D tensor).
__device__ __forceinline__ void tdm_load_2d(unsigned lds_byte,
                                            const __bf16* gptr,
                                            unsigned tile_d0, unsigned tile_d1,
                                            unsigned tensor_d0, unsigned tensor_d1,
                                            unsigned stride0) {
  unsigned long long ga = (unsigned long long)gptr;
  v4u g0;
  g0.x = 1u;                                             // count=1, user mode
  g0.y = lds_byte;                                       // LDS byte address
  g0.z = (unsigned)(ga & 0xffffffffu);                   // global addr lo
  g0.w = (unsigned)((ga >> 32) & 0x01ffffffu) | (2u << 30); // hi + type=2
  v8u g1;
  g1[0] = 0x06D10000u;  // data_size=2B | pad_enable | interval=16DW | amount=4DW
  g1[1] = (tensor_d0 & 0xffffu) << 16;                   // atomic_barrier=0
  g1[2] = (tensor_d0 >> 16) | ((tensor_d1 & 0xffffu) << 16);
  g1[3] = (tensor_d1 >> 16) | (tile_d0 << 16);
  g1[4] = tile_d1;                                       // tile_dim2 = 0
  g1[5] = stride0;                                       // dim0 stride lo32
  g1[6] = 0u;
  g1[7] = 0u;
  asm volatile("tensor_load_to_lds %0, %1" ::"s"(g0), "s"(g1) : "memory");
}

// ---------------------------------------------------------------------------
// 0) f32 -> bf16 conversion
// ---------------------------------------------------------------------------
__global__ void k_cvt_bf16(const float* __restrict__ s, __bf16* __restrict__ d, int n) {
  int i = blockIdx.x * blockDim.x + threadIdx.x;
  if (i < n) d[i] = f2bf(s[i]);
}

// ---------------------------------------------------------------------------
// 1) Projection GEMM: C[4096][1024] = A @ W^T. 8 waves, 128x64 macro-tile,
//    TDM double-buffered LDS staging, K-step 32.
//    transposed==0: store [b,h,s,hd] (Q scaled, K);  ==1: store [b,h,hd,s].
//    Grid: (MROWS/128)*(HIDDEN/64) = 512 blocks of 256 threads (exact).
// ---------------------------------------------------------------------------
__global__ void k_proj(const __bf16* __restrict__ A, const __bf16* __restrict__ W,
                       __bf16* __restrict__ out, float scale, int transposed) {
  __shared__ __align__(16) __bf16 sA[2][128 * LROW32];
  __shared__ __align__(16) __bf16 sB[2][64 * LROW32];
  int wid = threadIdx.x >> 5, lane = threadIdx.x & 31;
  int l = lane & 15, hi = lane >> 4;
  int bm = blockIdx.x >> 4, bn = blockIdx.x & 15;    // 32 x 16 macro-tiles
  int m0 = bm * 128, n0 = bn * 64;

  if (wid == 0) {
    tdm_load_2d(lds_off(sA[0]), A + (size_t)m0 * DIM, 32, 128, DIM, MROWS, DIM);
    tdm_load_2d(lds_off(sB[0]), W + (size_t)n0 * DIM, 32, 64, DIM, HIDDEN, DIM);
  }
  v8f acc[4] = {};
  for (int k0 = 0; k0 < DIM; k0 += 32) {
    int cur = (k0 >> 5) & 1;
    if (wid == 0) {
      if (k0 + 32 < DIM) {
        tdm_load_2d(lds_off(sA[cur ^ 1]), A + (size_t)m0 * DIM + k0 + 32,
                    32, 128, DIM, MROWS, DIM);
        tdm_load_2d(lds_off(sB[cur ^ 1]), W + (size_t)n0 * DIM + k0 + 32,
                    32, 64, DIM, HIDDEN, DIM);
        WAIT_TENSOR(2);                      // current pair done, next in flight
      } else {
        WAIT_TENSOR(0);
      }
    }
    __syncthreads();
    v16bf af = load_frag16(&sA[cur][(wid * 16 + l) * LROW32], hi);
#pragma unroll
    for (int nt = 0; nt < 4; ++nt)
      acc[nt] = wmma_bf16(af, load_frag16(&sB[cur][(nt * 16 + l) * LROW32], hi),
                          acc[nt]);
    __syncthreads();
  }
  int gm0 = m0 + wid * 16;
  int b = gm0 >> 11, i0 = gm0 & (SEQ - 1);
#pragma unroll
  for (int nt = 0; nt < 4; ++nt) {
    int n = n0 + nt * 16 + l;
    int h = n >> 6, d = n & 63;
    if (!transposed) {
      __bf16* p = out + (((size_t)b * N_HEADS + h) * SEQ + i0 + hi * 8) * HD + d;
#pragma unroll
      for (int r = 0; r < 8; ++r) p[(size_t)r * HD] = f2bf(acc[nt][r] * scale);
    } else {
      __bf16* p = out + (((size_t)b * N_HEADS + h) * HD + d) * SEQ + i0 + hi * 8;
#pragma unroll
      for (int r = 0; r < 8; ++r) p[r] = f2bf(acc[nt][r] * scale);
    }
  }
}

// ---------------------------------------------------------------------------
// 2) Scores: S[h,b,i,j] = (Q*SCALE) @ K^T, masked; per-row max.
//    8 waves/block share one (b,h); each K 16x64 j-tile TDM-staged once.
//    Grid: 512 blocks of 256 threads (exact).
// ---------------------------------------------------------------------------
__global__ void k_scores(const __bf16* __restrict__ Q, const __bf16* __restrict__ K,
                         const int* __restrict__ mask, float* __restrict__ attn,
                         float* __restrict__ rowmax) {
  __shared__ __align__(16) __bf16 sK[2][16 * LROW64];
  int wid = threadIdx.x >> 5, lane = threadIdx.x & 31;
  int l = lane & 15, hi = lane >> 4;
  int wave = blockIdx.x * 8 + wid;
  int bh = wave >> 7, it = wave & 127;      // 16 blocks per (b,h): uniform bh
  int b = bh >> 4, h = bh & 15;

  const __bf16* Qb = Q + (size_t)bh * SEQ * HD;
  const __bf16* Kb = K + (size_t)bh * SEQ * HD;
  v16bf qa0 = load_frag16(Qb + (size_t)(it * 16 + l) * HD, hi);       // K 0..31
  v16bf qa1 = load_frag16(Qb + (size_t)(it * 16 + l) * HD + 32, hi);  // K 32..63

  float rmax[8];
#pragma unroll
  for (int r = 0; r < 8; ++r) rmax[r] = BIGNEG;
  float* arow = attn + ((size_t)(h * BATCH + b) * SEQ + it * 16 + hi * 8) * SEQ;

  if (wid == 0) tdm_load_2d(lds_off(sK[0]), Kb, 64, 16, HD, SEQ, HD);

  for (int jt = 0; jt < SEQ / 16; ++jt) {
    int cur = jt & 1;
    if (wid == 0) {
      if (jt + 1 < SEQ / 16) {
        tdm_load_2d(lds_off(sK[cur ^ 1]), Kb + (size_t)(jt + 1) * 16 * HD,
                    64, 16, HD, SEQ, HD);
        WAIT_TENSOR(1);
      } else {
        WAIT_TENSOR(0);
      }
    }
    __syncthreads();
    const __bf16* kp = &sK[cur][l * LROW64];
    v8f s = {};
    s = wmma_bf16(qa0, load_frag16(kp, hi), s);           // K half 0
    s = wmma_bf16(qa1, load_frag16(kp + LROW32, hi), s);  // K half 1 (+40 elems)
    int j  = jt * 16 + l;
    int mv = mask[b * SEQ + j];
#pragma unroll
    for (int r = 0; r < 8; ++r) {
      float v = mv ? s[r] : BIGNEG;
      arow[(size_t)r * SEQ + j] = v;
      rmax[r] = fmaxf(rmax[r], v);
    }
    __syncthreads();
  }
#pragma unroll
  for (int off = 1; off < 16; off <<= 1)
#pragma unroll
    for (int r = 0; r < 8; ++r)
      rmax[r] = fmaxf(rmax[r], __shfl_xor(rmax[r], off, 32));
  if (l == 0) {
    float* pm = rowmax + (size_t)bh * SEQ + it * 16 + hi * 8;
#pragma unroll
    for (int r = 0; r < 8; ++r) pm[r] = rmax[r];
  }
}

// ---------------------------------------------------------------------------
// 3) In-place exp(s - rowmax) + row sums. One wave per attention row.
// ---------------------------------------------------------------------------
__global__ void k_softexp(float* __restrict__ attn, const float* __restrict__ rowmax,
                          float* __restrict__ rowsum) {
  int wave = blockIdx.x * (blockDim.x >> 5) + (threadIdx.x >> 5);
  int lane = threadIdx.x & 31;
  if (wave >= BATCH * N_HEADS * SEQ) return;
  int bh = wave >> 11;
  int i  = wave & (SEQ - 1);
  int b = bh >> 4, h = bh & 15;
  float m = rowmax[(size_t)bh * SEQ + i];
  float* row = attn + ((size_t)(h * BATCH + b) * SEQ + i) * SEQ;
  float sum = 0.f;
  for (int j = lane; j < SEQ; j += 32) {
    float e = __expf(row[j] - m);
    row[j] = e;
    sum += e;
  }
#pragma unroll
  for (int off = 1; off < 32; off <<= 1) sum += __shfl_xor(sum, off, 32);
  if (lane == 0) rowsum[(size_t)bh * SEQ + i] = sum;
}

// ---------------------------------------------------------------------------
// 4) Normalize attn in place (final attn output) and O = attn @ V.
//    8 waves/block share one (b,h); V^T 64x32 tiles TDM-staged once per block.
//    Grid: 512 blocks of 256 threads (exact).
// ---------------------------------------------------------------------------
__global__ void k_pv(float* __restrict__ attn, const float* __restrict__ rowsum,
                     const __bf16* __restrict__ Vt, __bf16* __restrict__ Oc) {
  __shared__ __align__(16) __bf16 sV[2][64 * LROW32];
  int wid = threadIdx.x >> 5, lane = threadIdx.x & 31;
  int l = lane & 15, hi = lane >> 4;
  int wave = blockIdx.x * 8 + wid;
  int bh = wave >> 7, it = wave & 127;
  int b = bh >> 4, h = bh & 15;

  float inv = 1.0f / rowsum[(size_t)bh * SEQ + it * 16 + l];
  float* arow = attn + ((size_t)(h * BATCH + b) * SEQ + it * 16 + l) * SEQ;
  const __bf16* Vb = Vt + (size_t)bh * HD * SEQ;      // [64][2048]

  if (wid == 0) tdm_load_2d(lds_off(sV[0]), Vb, 32, 64, SEQ, HD, SEQ);

  v8f acc[4] = {};
  for (int j0 = 0; j0 < SEQ; j0 += 32) {
    int cur = (j0 >> 5) & 1;
    if (wid == 0) {
      if (j0 + 32 < SEQ) {
        tdm_load_2d(lds_off(sV[cur ^ 1]), Vb + j0 + 32, 32, 64, SEQ, HD, SEQ);
        WAIT_TENSOR(1);
      } else {
        WAIT_TENSOR(0);
      }
    }
    __syncthreads();
    FragBF pf;
#pragma unroll
    for (int r = 0; r < 8; ++r) {
      float e0 = arow[j0 + hi * 8 + r] * inv;          // K = j0 + hi*8 + r
      arow[j0 + hi * 8 + r] = e0;                      // final normalized attn
      pf.v[r] = f2bf(e0);
      float e1 = arow[j0 + 16 + hi * 8 + r] * inv;     // K = j0 + 16 + hi*8 + r
      arow[j0 + 16 + hi * 8 + r] = e1;
      pf.v[8 + r] = f2bf(e1);
    }
#pragma unroll
    for (int nt = 0; nt < 4; ++nt)
      acc[nt] = wmma_bf16(pf.v,
                          load_frag16(&sV[cur][(nt * 16 + l) * LROW32], hi),
                          acc[nt]);
    __syncthreads();
  }
#pragma unroll
  for (int nt = 0; nt < 4; ++nt) {
    size_t base = ((size_t)b * SEQ + it * 16 + hi * 8) * HIDDEN + h * HD + nt * 16 + l;
#pragma unroll
    for (int r = 0; r < 8; ++r) Oc[base + (size_t)r * HIDDEN] = f2bf(acc[nt][r]);
  }
}

// ---------------------------------------------------------------------------
// 5) Output projection: out[4096][1024] f32 = Oc @ Wo^T (TDM-staged LDS).
//    Grid: 512 blocks of 256 threads (exact).
// ---------------------------------------------------------------------------
__global__ void k_outproj(const __bf16* __restrict__ Oc, const __bf16* __restrict__ Wo,
                          float* __restrict__ out) {
  __shared__ __align__(16) __bf16 sA[2][128 * LROW32];
  __shared__ __align__(16) __bf16 sB[2][64 * LROW32];
  int wid = threadIdx.x >> 5, lane = threadIdx.x & 31;
  int l = lane & 15, hi = lane >> 4;
  int bm = blockIdx.x >> 4, bn = blockIdx.x & 15;
  int m0 = bm * 128, n0 = bn * 64;

  if (wid == 0) {
    tdm_load_2d(lds_off(sA[0]), Oc + (size_t)m0 * HIDDEN, 32, 128, HIDDEN, MROWS, HIDDEN);
    tdm_load_2d(lds_off(sB[0]), Wo + (size_t)n0 * HIDDEN, 32, 64, HIDDEN, DIM, HIDDEN);
  }
  v8f acc[4] = {};
  for (int k0 = 0; k0 < HIDDEN; k0 += 32) {
    int cur = (k0 >> 5) & 1;
    if (wid == 0) {
      if (k0 + 32 < HIDDEN) {
        tdm_load_2d(lds_off(sA[cur ^ 1]), Oc + (size_t)m0 * HIDDEN + k0 + 32,
                    32, 128, HIDDEN, MROWS, HIDDEN);
        tdm_load_2d(lds_off(sB[cur ^ 1]), Wo + (size_t)n0 * HIDDEN + k0 + 32,
                    32, 64, HIDDEN, DIM, HIDDEN);
        WAIT_TENSOR(2);
      } else {
        WAIT_TENSOR(0);
      }
    }
    __syncthreads();
    v16bf af = load_frag16(&sA[cur][(wid * 16 + l) * LROW32], hi);
#pragma unroll
    for (int nt = 0; nt < 4; ++nt)
      acc[nt] = wmma_bf16(af, load_frag16(&sB[cur][(nt * 16 + l) * LROW32], hi),
                          acc[nt]);
    __syncthreads();
  }
  float* p = out + (size_t)(m0 + wid * 16 + hi * 8) * DIM + n0 + l;
#pragma unroll
  for (int nt = 0; nt < 4; ++nt)
#pragma unroll
    for (int r = 0; r < 8; ++r) p[(size_t)r * DIM + nt * 16] = acc[nt][r];
}

// ---------------------------------------------------------------------------
extern "C" void kernel_launch(void* const* d_in, const int* in_sizes, int n_in,
                              void* d_out, int out_size, void* d_ws, size_t ws_size,
                              hipStream_t stream) {
  (void)in_sizes; (void)n_in; (void)out_size; (void)ws_size;
  const float* x  = (const float*)d_in[0];
  const int* mask = (const int*)d_in[1];
  const float* Wq = (const float*)d_in[2];
  const float* Wk = (const float*)d_in[3];
  const float* Wv = (const float*)d_in[4];
  const float* Wo = (const float*)d_in[5];

  float* out  = (float*)d_out;                          // [b,s,DIM] f32
  float* attn = out + (size_t)MROWS * DIM;              // [h,b,i,j] f32

  __bf16* xb  = (__bf16*)d_ws;                          // [4096,1024]
  __bf16* wqb = xb  + (size_t)MROWS  * DIM;
  __bf16* wkb = wqb + (size_t)HIDDEN * DIM;
  __bf16* wvb = wkb + (size_t)HIDDEN * DIM;
  __bf16* wob = wvb + (size_t)HIDDEN * DIM;
  __bf16* qb  = wob + (size_t)DIM * HIDDEN;             // [b,h,s,64] (scaled)
  __bf16* kb  = qb  + (size_t)MROWS * HIDDEN;           // [b,h,s,64]
  __bf16* vtb = kb  + (size_t)MROWS * HIDDEN;           // [b,h,64,s]
  __bf16* ocb = vtb + (size_t)MROWS * HIDDEN;           // [b,s,1024]
  float* rowmax = (float*)(ocb + (size_t)MROWS * HIDDEN);
  float* rowsum = rowmax + (size_t)BATCH * N_HEADS * SEQ;

  const int T = 256;
  k_cvt_bf16<<<(MROWS * DIM) / T, T, 0, stream>>>(x, xb, MROWS * DIM);
  k_cvt_bf16<<<(HIDDEN * DIM) / T, T, 0, stream>>>(Wq, wqb, HIDDEN * DIM);
  k_cvt_bf16<<<(HIDDEN * DIM) / T, T, 0, stream>>>(Wk, wkb, HIDDEN * DIM);
  k_cvt_bf16<<<(HIDDEN * DIM) / T, T, 0, stream>>>(Wv, wvb, HIDDEN * DIM);
  k_cvt_bf16<<<(HIDDEN * DIM) / T, T, 0, stream>>>(Wo, wob, HIDDEN * DIM);

  int gemm_blocks = (MROWS / 128) * (HIDDEN / 64);      // 512, exact
  k_proj<<<gemm_blocks, T, 0, stream>>>(xb, wqb, qb, SCALE_Q, 0);
  k_proj<<<gemm_blocks, T, 0, stream>>>(xb, wkb, kb, 1.0f, 0);
  k_proj<<<gemm_blocks, T, 0, stream>>>(xb, wvb, vtb, 1.0f, 1);

  int att_blocks = (BATCH * N_HEADS * (SEQ / 16)) / 8;  // 512, exact
  k_scores<<<att_blocks, T, 0, stream>>>(qb, kb, mask, attn, rowmax);
  k_softexp<<<(BATCH * N_HEADS * SEQ) / 8, T, 0, stream>>>(attn, rowmax, rowsum);
  k_pv<<<att_blocks, T, 0, stream>>>(attn, rowsum, vtb, ocb);
  k_outproj<<<gemm_blocks, T, 0, stream>>>(ocb, wob, out);
}
